// MLPTexture3D_70153995812959
// MI455X (gfx1250) — matrix operand order repaired and checked by reading, not compiled, not run on hardware
//
#include <hip/hip_runtime.h>
#include <hip/hip_bf16.h>

// MLPTexture3D: tcnn-style HashGrid encode (16 levels, 2 feats, T=2^19, trilinear)
// followed by MLP 32->32 relu, 32->32 relu, 32->3 sigmoid.
//
// Mapping (wave32, CDNA5 WMMA 16x16x32 f16):
//   - each wave processes batches of 16 points
//   - lane L: point m = L&15; lanes 0-15 encode levels {0..7},
//     lanes 16-31 encode levels {8..15}  ->  the 16 f16 features per
//     lane are exactly the A-operand halves (K = kbase+h, contiguous,
//     kbase = (L>>4)*16), same K order as B. First A matrix built in registers.
//   - weights pre-loaded into B-operand layout (lane = out column N = L&15,
//     lanes 0-15 hold K=0..15, lanes 16-31 hold K=16..31, 2 halves/VGPR).
//   - layer transitions: D layout (lane=column, VGPR=row) -> LDS (1KB/wave)
//     -> reload in A layout with two ds_load_b128. Wave-private, in-order LDS;
//     explicit s_wait_dscnt between store/load phases.

typedef _Float16 v16h __attribute__((ext_vector_type(16)));
typedef _Float16 v8h  __attribute__((ext_vector_type(8)));
typedef float    v8f  __attribute__((ext_vector_type(8)));

#define LOG2_T      19
#define TABLE_SIZE  (1u << LOG2_T)
#define N_POINTS    4194304
#define BATCHES     (N_POINTS / 16)   // 262144

__device__ __forceinline__ float sigmoid_f(float x) {
    return 1.0f / (1.0f + __expf(-x));
}

__global__ __launch_bounds__(256) void MLPTexture3D_kernel(
    const float* __restrict__ texc,   // [N,3]
    const float* __restrict__ table,  // [16, 2^19, 2]
    const float* __restrict__ W0,     // [32,32]
    const float* __restrict__ W1,     // [32,32]
    const float* __restrict__ W2,     // [3,32]
    float* __restrict__ out)          // [N,3]
{
    __shared__ __align__(16) _Float16 smem[8 * 512];   // 1KB (16pt x 32feat f16) per wave
    const int lane      = threadIdx.x & 31;
    const int waveInBlk = threadIdx.x >> 5;
    _Float16* my = &smem[waveInBlk * 512];

    const int hi    = lane >> 4;     // half-wave id
    const int m     = lane & 15;     // A row (point) / B,C,D column
    const int kbase = hi * 16;       // A operand K base (halves)
    const int ksel  = hi * 16;       // B operand K base

    // ---- weights, resident in registers in WMMA B layout --------------------
    v16h b0w0, b1w0, b0w1, b1w1, bw2;
    #pragma unroll
    for (int h = 0; h < 16; ++h) {
        b0w0[h] = (_Float16)W0[(m     ) * 32 + ksel + h];
        b1w0[h] = (_Float16)W0[(m + 16) * 32 + ksel + h];
        b0w1[h] = (_Float16)W1[(m     ) * 32 + ksel + h];
        b1w1[h] = (_Float16)W1[(m + 16) * 32 + ksel + h];
        bw2[h]  = (m < 3) ? (_Float16)W2[m * 32 + ksel + h] : (_Float16)0.0f;
    }

    // ---- per-lane level set (8 levels per lane) -----------------------------
    uint  lvl_j[8];
    float scale_j[8];
    uint  res_j[8], res2_j[8];
    #pragma unroll
    for (int j = 0; j < 8; ++j) {
        uint lvl = (uint)(hi * 8 + j);   // {0..7} / {8..15}
        lvl_j[j] = lvl;
        // scale = exp2(lvl * log2(growth)) * 16 - 1, log2(growth) = 8/15
        float scale = exp2f((float)lvl * (8.0f / 15.0f)) * 16.0f - 1.0f;
        scale_j[j]  = scale;
        uint res    = (uint)ceilf(scale) + 1u;
        res_j[j]    = res;
        res2_j[j]   = res * res;
    }

    const int nwaves = (gridDim.x * blockDim.x) >> 5;
    const int gwave  = (blockIdx.x * blockDim.x + threadIdx.x) >> 5;

    for (int batch = gwave; batch < BATCHES; batch += nwaves) {
        const int p = batch * 16 + m;              // this lane's point
        const float tx = texc[p * 3 + 0];
        const float ty = texc[p * 3 + 1];
        const float tz = texc[p * 3 + 2];

        // ---- hashgrid encode: builds the A operand directly in registers ----
        v16h A;
        #pragma unroll
        for (int j = 0; j < 8; ++j) {
            const uint  lvl   = lvl_j[j];
            const float scale = scale_j[j];
            const float px = tx * scale + 0.5f;
            const float py = ty * scale + 0.5f;
            const float pz = tz * scale + 0.5f;
            const float gx = floorf(px), gy = floorf(py), gz = floorf(pz);
            const float fx = px - gx, fy = py - gy, fz = pz - gz;
            const uint  bx = (uint)gx, by = (uint)gy, bz = (uint)gz;
            const float2* tl = (const float2*)table + ((size_t)lvl << LOG2_T);
            const bool dense = (lvl < 5u);         // levels 0..4: res^3 <= 2^19
            float f0 = 0.0f, f1 = 0.0f;
            #pragma unroll
            for (int c = 0; c < 8; ++c) {
                const uint dx = c & 1, dy = (c >> 1) & 1, dz = c >> 2;
                const uint cx = bx + dx, cy = by + dy, cz = bz + dz;
                const float w = (dx ? fx : 1.0f - fx)
                              * (dy ? fy : 1.0f - fy)
                              * (dz ? fz : 1.0f - fz);
                uint idx;
                if (dense) idx = cx + cy * res_j[j] + cz * res2_j[j];
                else       idx = (cx ^ (cy * 2654435761u) ^ (cz * 805459861u))
                                 & (TABLE_SIZE - 1u);
                const float2 g = tl[idx];          // global_load_b64, L2-resident
                f0 += w * g.x;
                f1 += w * g.y;
            }
            A[2 * j]     = (_Float16)f0;
            A[2 * j + 1] = (_Float16)f1;
        }

        // ---- layer 0: h = relu(E @ W0^T), two 16x16 output tiles ------------
        v8f c0 = {}, c1 = {};
        c0 = __builtin_amdgcn_wmma_f32_16x16x32_f16(false, A, false, b0w0,
                                                    (short)0, c0, false, false);
        c1 = __builtin_amdgcn_wmma_f32_16x16x32_f16(false, A, false, b1w0,
                                                    (short)0, c1, false, false);

        // D layout -> LDS (row-major [pt][feat] f16), relu fused
        #pragma unroll
        for (int v = 0; v < 8; ++v) {
            const int pt = v + 8 * hi;
            const float r0 = c0[v] > 0.0f ? c0[v] : 0.0f;
            const float r1 = c1[v] > 0.0f ? c1[v] : 0.0f;
            my[pt * 32 + m]      = (_Float16)r0;
            my[pt * 32 + m + 16] = (_Float16)r1;
        }
        asm volatile("s_wait_dscnt 0" ::: "memory");
        v8h lo  = *(const v8h*)&my[m * 32 + kbase];        // ds_load_b128
        v8h hi8 = *(const v8h*)&my[m * 32 + 8 + kbase];    // ds_load_b128
        v16h A1 = __builtin_shufflevector(lo, hi8,
                    0,1,2,3,4,5,6,7,8,9,10,11,12,13,14,15);

        // ---- layer 1 --------------------------------------------------------
        v8f d0 = {}, d1 = {};
        d0 = __builtin_amdgcn_wmma_f32_16x16x32_f16(false, A1, false, b0w1,
                                                    (short)0, d0, false, false);
        d1 = __builtin_amdgcn_wmma_f32_16x16x32_f16(false, A1, false, b1w1,
                                                    (short)0, d1, false, false);
        #pragma unroll
        for (int v = 0; v < 8; ++v) {
            const int pt = v + 8 * hi;
            const float r0 = d0[v] > 0.0f ? d0[v] : 0.0f;
            const float r1 = d1[v] > 0.0f ? d1[v] : 0.0f;
            my[pt * 32 + m]      = (_Float16)r0;
            my[pt * 32 + m + 16] = (_Float16)r1;
        }
        asm volatile("s_wait_dscnt 0" ::: "memory");
        lo  = *(const v8h*)&my[m * 32 + kbase];
        hi8 = *(const v8h*)&my[m * 32 + 8 + kbase];
        v16h A2 = __builtin_shufflevector(lo, hi8,
                    0,1,2,3,4,5,6,7,8,9,10,11,12,13,14,15);

        // ---- layer 2: 3 sigmoid outputs (B padded to 16 cols, 3 valid) ------
        v8f e0 = {};
        e0 = __builtin_amdgcn_wmma_f32_16x16x32_f16(false, A2, false, bw2,
                                                    (short)0, e0, false, false);
        if (m < 3) {
            #pragma unroll
            for (int v = 0; v < 8; ++v) {
                const int pt = batch * 16 + v + 8 * hi;
                out[pt * 3 + m] = sigmoid_f(e0[v]);
            }
        }
    }
}

extern "C" void kernel_launch(void* const* d_in, const int* in_sizes, int n_in,
                              void* d_out, int out_size, void* d_ws, size_t ws_size,
                              hipStream_t stream) {
    (void)in_sizes; (void)n_in; (void)out_size; (void)d_ws; (void)ws_size;
    const float* texc  = (const float*)d_in[0];
    const float* table = (const float*)d_in[1];
    const float* W0    = (const float*)d_in[2];
    const float* W1    = (const float*)d_in[3];
    const float* W2    = (const float*)d_in[4];
    float* out = (float*)d_out;

    // 2048 blocks x 256 threads = 16384 waves -> 16 batches of 16 points each.
    dim3 grid(2048), block(256);
    MLPTexture3D_kernel<<<grid, block, 0, stream>>>(texc, table, W0, W1, W2, out);
}
